// MambaBlock_18124761989161
// MI455X (gfx1250) — compile-verified, matrix operand order checked by reading, not run on hardware
//
#include <hip/hip_runtime.h>
#include <hip/hip_bf16.h>

typedef __bf16 bf16;
typedef __attribute__((ext_vector_type(16))) __bf16 v16bf;
typedef __attribute__((ext_vector_type(8)))  __bf16 v8bf;
typedef __attribute__((ext_vector_type(8)))  float  v8f;

#define D_MODEL 1024
#define D_INNER 2048
#define D_STATE 16
#define DT_RANK 64
#define SEQ_B   2
#define SEQ_L   2048
#define NROWS   (SEQ_B*SEQ_L)   // 4096 token rows
#define XZ_N    (2*D_INNER)     // 4096
#define XDBL_N  (DT_RANK + 2*D_STATE) // 96

// ---------------------------------------------------------------------------
// fp32 -> bf16 weight conversion
// ---------------------------------------------------------------------------
__global__ void cvt_bf16_kernel(const float* __restrict__ src, bf16* __restrict__ dst, int n) {
    int i = blockIdx.x * blockDim.x + threadIdx.x;
    if (i < n) dst[i] = (bf16)src[i];
}

// ---------------------------------------------------------------------------
// LayerNorm (D=1024) -> bf16 rows (A-matrix for in_proj WMMA GEMM)
// one block (256 thr) per token row
// ---------------------------------------------------------------------------
__global__ void ln_bf16_kernel(const float* __restrict__ x, const float* __restrict__ w,
                               const float* __restrict__ b, bf16* __restrict__ h) {
    const int D = D_MODEL;
    int row = blockIdx.x, tid = threadIdx.x;
    const float4 v = ((const float4*)(x + (size_t)row * D))[tid];
    __shared__ float rs[256], rss[256];
    rs[tid]  = v.x + v.y + v.z + v.w;
    rss[tid] = v.x*v.x + v.y*v.y + v.z*v.z + v.w*v.w;
    __syncthreads();
    for (int off = 128; off > 0; off >>= 1) {
        if (tid < off) { rs[tid] += rs[tid+off]; rss[tid] += rss[tid+off]; }
        __syncthreads();
    }
    float mu  = rs[0]  * (1.0f / D);
    float var = rss[0] * (1.0f / D) - mu * mu;
    float inv = rsqrtf(var + 1e-6f);
    int c = tid * 4;
    bf16* hr = h + (size_t)row * D + c;
    hr[0] = (bf16)((v.x - mu) * inv * w[c+0] + b[c+0]);
    hr[1] = (bf16)((v.y - mu) * inv * w[c+1] + b[c+1]);
    hr[2] = (bf16)((v.z - mu) * inv * w[c+2] + b[c+2]);
    hr[3] = (bf16)((v.w - mu) * inv * w[c+3] + b[c+3]);
}

// ---------------------------------------------------------------------------
// bf16 WMMA GEMM:  C[m,n] = sum_k A[m,k] * Bw[n,k]  (+ optional residual)
// A: MxK row-major bf16, Bw: NxK row-major bf16, C: MxN fp32.
// Wave computes a 32x64 tile: 2 A-frags + 4 B-frags -> 8 WMMAs per K=32 step
// (1.5 b128 loads per WMMA).
// Fragment layout per ISA 7.12.2: lane<16 holds K {0..7,16..23},
// lane>=16 holds K {8..15,24..31}, row/col = lane&15.
// ---------------------------------------------------------------------------
__device__ __forceinline__ v16bf load_frag(const bf16* q) {
    union { v16bf v; v8bf h[2]; } f;
    f.h[0] = *(const v8bf*)(q);        // K = kh+0..7
    f.h[1] = *(const v8bf*)(q + 16);   // K = kh+16..23
    return f.v;
}

__global__ void gemm_bf16_wmma_kernel(const bf16* __restrict__ A, const bf16* __restrict__ Bw,
                                      const float* __restrict__ Res, float* __restrict__ C,
                                      int N, int K) {
    const int lane = threadIdx.x & 31;
    const int wave = threadIdx.x >> 5;
    const int m0 = blockIdx.y * 128 + wave * 32;   // 32 rows per wave
    const int n0 = blockIdx.x * 64;                // 64 cols per wave
    const int r  = lane & 15;
    const int kh = (lane >> 4) << 3;               // 0 or 8
    const bf16* a0 = A  + (size_t)(m0 + r) * K + kh;
    const bf16* a1 = a0 + (size_t)16 * K;
    const bf16* b  = Bw + (size_t)(n0 + r) * K + kh;
    const size_t bstep = (size_t)16 * K;
    v8f acc[8];
    #pragma unroll
    for (int i = 0; i < 8; ++i) acc[i] = (v8f){};
    for (int k = 0; k < K; k += 32) {
        __builtin_prefetch(a0 + k + 128, 0, 3);    // global_prefetch_b8, near-cache
        __builtin_prefetch(b  + k + 128, 0, 3);
        v16bf af0 = load_frag(a0 + k);
        v16bf af1 = load_frag(a1 + k);
        v16bf bf0 = load_frag(b + k);
        v16bf bf1 = load_frag(b + bstep     + k);
        v16bf bf2 = load_frag(b + 2 * bstep + k);
        v16bf bf3 = load_frag(b + 3 * bstep + k);
        acc[0] = __builtin_amdgcn_wmma_f32_16x16x32_bf16(false, af0, false, bf0, (short)0, acc[0], false, false);
        acc[1] = __builtin_amdgcn_wmma_f32_16x16x32_bf16(false, af0, false, bf1, (short)0, acc[1], false, false);
        acc[2] = __builtin_amdgcn_wmma_f32_16x16x32_bf16(false, af0, false, bf2, (short)0, acc[2], false, false);
        acc[3] = __builtin_amdgcn_wmma_f32_16x16x32_bf16(false, af0, false, bf3, (short)0, acc[3], false, false);
        acc[4] = __builtin_amdgcn_wmma_f32_16x16x32_bf16(false, af1, false, bf0, (short)0, acc[4], false, false);
        acc[5] = __builtin_amdgcn_wmma_f32_16x16x32_bf16(false, af1, false, bf1, (short)0, acc[5], false, false);
        acc[6] = __builtin_amdgcn_wmma_f32_16x16x32_bf16(false, af1, false, bf2, (short)0, acc[6], false, false);
        acc[7] = __builtin_amdgcn_wmma_f32_16x16x32_bf16(false, af1, false, bf3, (short)0, acc[7], false, false);
    }
    // C layout: VGPR i -> row base + i + (lane>=16?8:0), col n0+(lane&15)+16*tile
    const int rowoff = (lane >> 4) * 8;
    const int col = n0 + (lane & 15);
    #pragma unroll
    for (int g = 0; g < 2; ++g) {
        #pragma unroll
        for (int i = 0; i < 8; ++i) {
            int row = m0 + g * 16 + rowoff + i;
            size_t o = (size_t)row * N + col;
            float r0 = acc[g*4+0][i], r1 = acc[g*4+1][i], r2 = acc[g*4+2][i], r3 = acc[g*4+3][i];
            if (Res) { r0 += Res[o]; r1 += Res[o+16]; r2 += Res[o+32]; r3 += Res[o+48]; }
            C[o] = r0; C[o+16] = r1; C[o+32] = r2; C[o+48] = r3;
        }
    }
}

// ---------------------------------------------------------------------------
// depthwise causal conv(4) + bias + SiLU.  xi = xz[:, 0:2048]
// ---------------------------------------------------------------------------
__global__ void conv_silu_kernel(const float* __restrict__ xz, const float* __restrict__ cw,
                                 const float* __restrict__ cb, float* __restrict__ u) {
    int idx = blockIdx.x * blockDim.x + threadIdx.x;
    if (idx >= NROWS * D_INNER) return;
    int d = idx & (D_INNER - 1);
    int m = idx >> 11;
    int l = m & (SEQ_L - 1);
    float acc = cb[d];
    #pragma unroll
    for (int j = 0; j < 4; ++j) {
        int li = l - 3 + j;
        if (li >= 0) acc += cw[d * 4 + j] * xz[(size_t)(m - 3 + j) * XZ_N + d];
    }
    u[idx] = acc / (1.0f + __expf(-acc)) ;
}

// ---------------------------------------------------------------------------
// x_proj: x_dbl[m,e] = sum_k u[m,k] * xw[e,k],  e < 96
// ---------------------------------------------------------------------------
__global__ void xproj_kernel(const float* __restrict__ u, const float* __restrict__ xw,
                             float* __restrict__ xdbl) {
    int idx = blockIdx.x * blockDim.x + threadIdx.x;
    if (idx >= NROWS * XDBL_N) return;
    int e = idx % XDBL_N;
    int m = idx / XDBL_N;
    const float4* ur = (const float4*)(u  + (size_t)m * D_INNER);
    const float4* wr = (const float4*)(xw + (size_t)e * D_INNER);
    float acc = 0.f;
    for (int i = 0; i < D_INNER / 4; ++i) {
        float4 a = ur[i], b = wr[i];
        acc += a.x*b.x + a.y*b.y + a.z*b.z + a.w*b.w;
    }
    xdbl[idx] = acc;
}

// ---------------------------------------------------------------------------
// delta = softplus(dt @ dt_proj_w^T + dt_proj_b), dt = x_dbl[:, :64]
// ---------------------------------------------------------------------------
__global__ void delta_kernel(const float* __restrict__ xdbl, const float* __restrict__ dtw,
                             const float* __restrict__ dtb, float* __restrict__ delta) {
    int idx = blockIdx.x * blockDim.x + threadIdx.x;
    if (idx >= NROWS * D_INNER) return;
    int d = idx & (D_INNER - 1);
    int m = idx >> 11;
    const float4* dtr = (const float4*)(xdbl + (size_t)m * XDBL_N);
    const float4* wr  = (const float4*)(dtw  + (size_t)d * DT_RANK);
    float acc = dtb[d];
    #pragma unroll
    for (int i = 0; i < DT_RANK / 4; ++i) {
        float4 a = dtr[i], b = wr[i];
        acc += a.x*b.x + a.y*b.y + a.z*b.z + a.w*b.w;
    }
    delta[idx] = (acc > 20.f) ? acc : log1pf(__expf(acc));
}

// ---------------------------------------------------------------------------
// selective scan, fused: deltaA/deltaBu on the fly, + D-term, * silu(z),
// emits bf16 rows for out_proj GEMM. thread = (b,d); 16 states in VGPRs.
// B/C vectors staged through LDS per timestep.
// ---------------------------------------------------------------------------
__global__ void scan_kernel(const float* __restrict__ delta, const float* __restrict__ u,
                            const float* __restrict__ xdbl, const float* __restrict__ xz,
                            const float* __restrict__ Alog, const float* __restrict__ Dp,
                            bf16* __restrict__ ybf) {
    const int b = blockIdx.y;
    const int d = blockIdx.x * blockDim.x + threadIdx.x;
    float Ad[D_STATE], st[D_STATE];
    #pragma unroll
    for (int n = 0; n < D_STATE; ++n) {
        Ad[n] = -__expf(Alog[(size_t)d * D_STATE + n]);
        st[n] = 0.f;
    }
    const float Dd = Dp[d];
    __shared__ float sB[D_STATE], sC[D_STATE];
    for (int l = 0; l < SEQ_L; ++l) {
        const int m = b * SEQ_L + l;
        if (threadIdx.x < 2 * D_STATE) {
            float v = xdbl[(size_t)m * XDBL_N + DT_RANK + threadIdx.x];
            if (threadIdx.x < D_STATE) sB[threadIdx.x] = v;
            else                       sC[threadIdx.x - D_STATE] = v;
        }
        __syncthreads();
        const float dlt = delta[(size_t)m * D_INNER + d];
        const float uu  = u[(size_t)m * D_INNER + d];
        float y = 0.f;
        #pragma unroll
        for (int n = 0; n < D_STATE; ++n) {
            float dA = __expf(dlt * Ad[n]);
            st[n] = dA * st[n] + dlt * sB[n] * uu;
            y += st[n] * sC[n];
        }
        const float zv = xz[(size_t)m * XZ_N + D_INNER + d];
        const float gate = zv / (1.0f + __expf(-zv));
        ybf[(size_t)m * D_INNER + d] = (bf16)((y + uu * Dd) * gate);
        __syncthreads();
    }
}

// ---------------------------------------------------------------------------
extern "C" void kernel_launch(void* const* d_in, const int* in_sizes, int n_in,
                              void* d_out, int out_size, void* d_ws, size_t ws_size,
                              hipStream_t stream) {
    const float* x     = (const float*)d_in[0];
    const float* lnw   = (const float*)d_in[1];
    const float* lnb   = (const float*)d_in[2];
    const float* inW   = (const float*)d_in[3];   // (4096, 1024)
    const float* convW = (const float*)d_in[4];   // (2048, 1, 4)
    const float* convB = (const float*)d_in[5];
    const float* xprojW= (const float*)d_in[6];   // (96, 2048)
    const float* dtW   = (const float*)d_in[7];   // (2048, 64)
    const float* dtB   = (const float*)d_in[8];
    const float* Alog  = (const float*)d_in[9];   // (2048, 16)
    const float* Dp    = (const float*)d_in[10];
    const float* outW  = (const float*)d_in[11];  // (1024, 2048)

    char* ws = (char*)d_ws;
    size_t off = 0;
    auto alloc = [&](size_t bytes) -> void* {
        void* p = ws + off;
        off = (off + bytes + 255) & ~(size_t)255;
        return p;
    };
    bf16*  hbf    = (bf16*) alloc((size_t)NROWS * D_MODEL * 2);   // LN output, bf16
    bf16*  inWbf  = (bf16*) alloc((size_t)XZ_N * D_MODEL * 2);    // in_proj_w bf16
    bf16*  outWbf = (bf16*) alloc((size_t)D_MODEL * D_INNER * 2); // out_proj_w bf16
    float* xz     = (float*)alloc((size_t)NROWS * XZ_N * 4);
    float* u      = (float*)alloc((size_t)NROWS * D_INNER * 4);
    float* xdbl   = (float*)alloc((size_t)NROWS * XDBL_N * 4);
    float* delta  = (float*)alloc((size_t)NROWS * D_INNER * 4);
    bf16*  ybf    = (bf16*) alloc((size_t)NROWS * D_INNER * 2);

    // 1) weight conversions
    {
        int n1 = XZ_N * D_MODEL;
        cvt_bf16_kernel<<<(n1 + 255) / 256, 256, 0, stream>>>(inW, inWbf, n1);
        int n2 = D_MODEL * D_INNER;
        cvt_bf16_kernel<<<(n2 + 255) / 256, 256, 0, stream>>>(outW, outWbf, n2);
    }
    // 2) LayerNorm -> bf16
    ln_bf16_kernel<<<NROWS, 256, 0, stream>>>(x, lnw, lnb, hbf);
    // 3) in_proj GEMM (WMMA): xz = h @ in_proj_w^T   M=4096 N=4096 K=1024
    gemm_bf16_wmma_kernel<<<dim3(XZ_N / 64, NROWS / 128), 128, 0, stream>>>(
        hbf, inWbf, nullptr, xz, XZ_N, D_MODEL);
    // 4) conv + SiLU -> u
    {
        int n = NROWS * D_INNER;
        conv_silu_kernel<<<(n + 255) / 256, 256, 0, stream>>>(xz, convW, convB, u);
    }
    // 5) x_proj -> x_dbl (dt | B | C)
    {
        int n = NROWS * XDBL_N;
        xproj_kernel<<<(n + 127) / 128, 128, 0, stream>>>(u, xprojW, xdbl);
    }
    // 6) dt_proj + softplus -> delta
    {
        int n = NROWS * D_INNER;
        delta_kernel<<<(n + 255) / 256, 256, 0, stream>>>(xdbl, dtW, dtB, delta);
    }
    // 7) fused selective scan (+D-term, *silu(z)) -> ybf
    scan_kernel<<<dim3(D_INNER / 256, SEQ_B), 256, 0, stream>>>(
        delta, u, xdbl, xz, Alog, Dp, ybf);
    // 8) out_proj GEMM (WMMA) + residual: out = y @ out_proj_w^T + x
    gemm_bf16_wmma_kernel<<<dim3(D_MODEL / 64, NROWS / 128), 128, 0, stream>>>(
        ybf, outWbf, x, (float*)d_out, D_MODEL, D_INNER);
}
